// AttentionHead_60301340836000
// MI455X (gfx1250) — matrix operand order repaired and checked by reading, not compiled, not run on hardware
//
#include <hip/hip_runtime.h>

typedef _Float16 v16h __attribute__((ext_vector_type(16)));
typedef _Float16 v8h  __attribute__((ext_vector_type(8)));
typedef float    v8f  __attribute__((ext_vector_type(8)));
typedef float    v4f  __attribute__((ext_vector_type(4)));
typedef int      v4i  __attribute__((ext_vector_type(4)));
typedef int      v8i  __attribute__((ext_vector_type(8)));

#define HS     64
#define CDIM   1024
#define TDIM   4096
#define BDIM   4
#define BT     (BDIM * TDIM)

static __device__ __forceinline__ v8f wmma16(v16h a, v16h b, v8f c) {
  // D = A(16x32 f16) * B(32x16 f16) + C(16x16 f32)
  return __builtin_amdgcn_wmma_f32_16x16x32_f16(false, a, false, b, (short)0, c,
                                                false, false);
}

static __device__ __forceinline__ v16h cat8(v8h lo, v8h hi) {
  v16h r;
#pragma unroll
  for (int j = 0; j < 8; ++j) { r[j] = lo[j]; r[j + 8] = hi[j]; }
  return r;
}

static __device__ __forceinline__ void wait_tensorcnt0() {
#if __has_builtin(__builtin_amdgcn_s_wait_tensorcnt)
  __builtin_amdgcn_s_wait_tensorcnt(0);
#else
  asm volatile("s_wait_tensorcnt 0x0" ::: "memory");
#endif
}

// TDM: async DMA of a [rows x 64] f16 tile (row-major, stride 64) into LDS.
static __device__ __forceinline__ void tdm_load_tile_f16_64w(
    const _Float16* gsrc, _Float16* lds_dst, int rows) {
  const unsigned long long ga = (unsigned long long)(uintptr_t)gsrc;
  const unsigned int ldsa = (unsigned int)(uintptr_t)lds_dst;  // low 32 = LDS off
  v4i g0;
  g0[0] = 1;                                   // count=1 (valid), user mode
  g0[1] = (int)ldsa;                           // lds_addr (bytes)
  g0[2] = (int)(unsigned int)ga;               // global_addr[31:0]
  g0[3] = (int)((unsigned int)((ga >> 32) & 0x01FFFFFFu) | 0x80000000u);
                                               // global_addr[56:32] | type=2
  v8i g1;
  g1[0] = 0x00010000;                          // wg_mask=0, data_size=1 (2B)
  g1[1] = (int)(64u << 16);                    // tensor_dim0 = 64 (bits 63:48)
  g1[2] = (int)((unsigned int)rows << 16);     // tensor_dim1 (bits 95:80)
  g1[3] = (int)(64u << 16);                    // tile_dim0 = 64 (bits 127:112)
  g1[4] = rows;                                // tile_dim1 (bits 143:128)
  g1[5] = 64;                                  // tensor_dim0_stride = 64
  g1[6] = 0;
  g1[7] = 0;
  asm volatile("tensor_load_to_lds %0, %1" ::"s"(g0), "s"(g1) : "memory");
}

// -----------------------------------------------------------------------------
// Kernel 1: K = x @ Wk, V = x @ Wv, stored as f16 [BT, HS] for WMMA use.
// Block = 256 threads (8 waves). Each wave: one 16-row M tile, all 4 N tiles.
// W tiles (32x64) staged transposed in LDS per K-step -> contiguous B frags.
// -----------------------------------------------------------------------------
__global__ __launch_bounds__(256) void proj_kv(
    const float* __restrict__ x, const float* __restrict__ Wk,
    const float* __restrict__ Wv, _Float16* __restrict__ K16,
    _Float16* __restrict__ V16) {
  __shared__ _Float16 sWkT[HS * 32];  // [n][k]
  __shared__ _Float16 sWvT[HS * 32];  // [n][k]

  const int tid = threadIdx.x;
  const int wave = tid >> 5;
  const int lane = tid & 31;
  const int half16 = lane >> 4;  // 0: lanes 0-15, 1: lanes 16-31
  const int l16 = lane & 15;
  const int mbase = blockIdx.x * 128 + wave * 16;

  v8f accK[4], accV[4];
#pragma unroll
  for (int n = 0; n < 4; ++n) { accK[n] = {}; accV[n] = {}; }

  // A-fragment source row (16-bit A 16x32 layout: lane=row, K split by half)
  const int arow = mbase + l16;
  const float* xrow = x + (size_t)arow * CDIM + (half16 ? 8 : 0);
  const int kb = half16 ? 8 : 0;  // B-frag K range per lane half

  for (int k0 = 0; k0 < CDIM; k0 += 32) {
    __syncthreads();
    {  // stage 32x64 W tiles transposed to f16 LDS; 8 elements per thread
      const int e = tid * 8;
      const int krow = e >> 6;
      const int ncol = e & 63;
      const size_t base = (size_t)(k0 + krow) * HS + ncol;
      v4f wk0 = *(const v4f*)(Wk + base);
      v4f wk1 = *(const v4f*)(Wk + base + 4);
      v4f wv0 = *(const v4f*)(Wv + base);
      v4f wv1 = *(const v4f*)(Wv + base + 4);
#pragma unroll
      for (int j = 0; j < 4; ++j) {
        sWkT[(ncol + j) * 32 + krow]     = (_Float16)wk0[j];
        sWkT[(ncol + 4 + j) * 32 + krow] = (_Float16)wk1[j];
        sWvT[(ncol + j) * 32 + krow]     = (_Float16)wv0[j];
        sWvT[(ncol + 4 + j) * 32 + krow] = (_Float16)wv1[j];
      }
    }
    __syncthreads();

    // A fragment: f32 -> f16 on the fly
    v4f a0 = *(const v4f*)(xrow + k0);
    v4f a1 = *(const v4f*)(xrow + k0 + 4);
    v4f a2 = *(const v4f*)(xrow + k0 + 16);
    v4f a3 = *(const v4f*)(xrow + k0 + 20);
    v16h af;
#pragma unroll
    for (int j = 0; j < 4; ++j) {
      af[j] = (_Float16)a0[j];      af[4 + j] = (_Float16)a1[j];
      af[8 + j] = (_Float16)a2[j];  af[12 + j] = (_Float16)a3[j];
    }

#pragma unroll
    for (int nt = 0; nt < 4; ++nt) {
      const int coln = nt * 16 + l16;
      v16h bk = cat8(*(const v8h*)(sWkT + coln * 32 + kb),
                     *(const v8h*)(sWkT + coln * 32 + kb + 16));
      accK[nt] = wmma16(af, bk, accK[nt]);
      v16h bv = cat8(*(const v8h*)(sWvT + coln * 32 + kb),
                     *(const v8h*)(sWvT + coln * 32 + kb + 16));
      accV[nt] = wmma16(af, bv, accV[nt]);
    }
  }

  // C/D layout: VGPR g -> row g (lanes 0-15) / g+8 (lanes 16-31), col = l16
#pragma unroll
  for (int nt = 0; nt < 4; ++nt)
#pragma unroll
    for (int g = 0; g < 8; ++g) {
      const int row = mbase + g + (half16 ? 8 : 0);
      const int col = nt * 16 + l16;
      K16[(size_t)row * HS + col] = (_Float16)accK[nt][g];
      V16[(size_t)row * HS + col] = (_Float16)accV[nt][g];
    }
}

// -----------------------------------------------------------------------------
// Kernel 2: fused causal flash attention. Q == K (faithful to the source bug).
// Block = 256 threads = 8 waves, one block per (batch, 128 query rows).
// K tile DMA'd by the Tensor Data Mover (TENSORcnt); V staged transposed by
// the ALU lanes in parallel with the DMA. Online softmax, all GEMMs via WMMA.
// -----------------------------------------------------------------------------
__global__ __launch_bounds__(256) void flash_attn(
    const _Float16* __restrict__ K16, const _Float16* __restrict__ V16,
    float* __restrict__ out) {
  __shared__ _Float16 sK[32 * HS];       // [s][h] (filled by TDM)
  __shared__ _Float16 sVt[HS * 32];      // [h][s]
  __shared__ _Float16 sP[8 * 16 * 32];   // per-wave P tile [r][s]

  const int tid = threadIdx.x;
  const int wave = tid >> 5;
  const int lane = tid & 31;
  const int half16 = lane >> 4;
  const int l16 = lane & 15;

  const int b = blockIdx.x >> 5;      // TDIM/128 = 32 q-blocks per batch
  const int qb = blockIdx.x & 31;
  const int bt0 = b * TDIM;           // flat row offset of this batch
  const int qbase = qb * 128 + wave * 16;  // query tile base (within batch)
  const int row_off = half16 ? 8 : 0;      // C/D row-half offset

  // Q fragments (two K=32 chunks over H=64), kept in registers
  v16h qf[2];
  {
    const _Float16* qp =
        K16 + (size_t)(bt0 + qbase + l16) * HS + (half16 ? 8 : 0);
#pragma unroll
    for (int hc = 0; hc < 2; ++hc)
      qf[hc] = cat8(*(const v8h*)(qp + hc * 32),
                    *(const v8h*)(qp + hc * 32 + 16));
  }

  v8f acc[4];
#pragma unroll
  for (int nt = 0; nt < 4; ++nt) acc[nt] = {};
  float m_i[8], l_i[8];
#pragma unroll
  for (int g = 0; g < 8; ++g) { m_i[g] = -1.0e30f; l_i[g] = 0.0f; }

  _Float16* myP = sP + wave * (16 * 32);
  const int nkb = (qb + 1) * 4;  // 32-key blocks needed under causal mask

  for (int kb = 0; kb < nkb; ++kb) {
    const int kbase = kb * 32;
    __syncthreads();
    // K tile: async tensor DMA (wave 0 issues; overlaps with V staging below)
    if (wave == 0)
      tdm_load_tile_f16_64w(K16 + (size_t)(bt0 + kbase) * HS, sK, 32);
    {  // V tile staged transposed into LDS; 8 halves per thread
      const int e = tid * 8;
      const int s = e >> 6;
      const int h = e & 63;
      v8h vv = *(const v8h*)(V16 + (size_t)(bt0 + kbase + s) * HS + h);
#pragma unroll
      for (int j = 0; j < 8; ++j) sVt[(h + j) * 32 + s] = vv[j];
      if (kb + 1 < nkb) {  // prefetch next key block
        __builtin_prefetch(
            (const void*)(K16 + (size_t)(bt0 + kbase + 32 + s) * HS + h), 0, 0);
        __builtin_prefetch(
            (const void*)(V16 + (size_t)(bt0 + kbase + 32 + s) * HS + h), 0, 0);
      }
    }
    if (wave == 0) wait_tensorcnt0();  // TDM done before anyone reads sK
    __syncthreads();

    // scores: S[16 x 32] via 2 column tiles x 2 h-chunks = 4 WMMAs
    v8f sc[2];
#pragma unroll
    for (int nt = 0; nt < 2; ++nt) {
      v8f c = {};
#pragma unroll
      for (int hc = 0; hc < 2; ++hc) {
        const _Float16* bp =
            sK + (nt * 16 + l16) * HS + hc * 32 + (half16 ? 8 : 0);
        v16h bf = cat8(*(const v8h*)bp, *(const v8h*)(bp + 16));
        c = wmma16(qf[hc], bf, c);
      }
      sc[nt] = c;
    }

    // online softmax (scale, causal mask, row reductions within lane halves)
#pragma unroll
    for (int g = 0; g < 8; ++g) {
      const int row = qbase + g + row_off;
      const int c0 = kbase + l16;
      float s0 = sc[0][g] * 0.125f;  // HEAD_SIZE^-0.5
      float s1 = sc[1][g] * 0.125f;
      if (c0 > row) s0 = -1.0e30f;
      if (c0 + 16 > row) s1 = -1.0e30f;
      float mx = fmaxf(s0, s1);
      mx = fmaxf(mx, __shfl_xor(mx, 1, 32));
      mx = fmaxf(mx, __shfl_xor(mx, 2, 32));
      mx = fmaxf(mx, __shfl_xor(mx, 4, 32));
      mx = fmaxf(mx, __shfl_xor(mx, 8, 32));
      const float mnew = fmaxf(m_i[g], mx);
      const float alpha = __expf(m_i[g] - mnew);
      const float p0 = __expf(s0 - mnew);
      const float p1 = __expf(s1 - mnew);
      float rs = p0 + p1;
      rs += __shfl_xor(rs, 1, 32);
      rs += __shfl_xor(rs, 2, 32);
      rs += __shfl_xor(rs, 4, 32);
      rs += __shfl_xor(rs, 8, 32);
      l_i[g] = l_i[g] * alpha + rs;
      m_i[g] = mnew;
#pragma unroll
      for (int nt = 0; nt < 4; ++nt) acc[nt][g] *= alpha;
      // spill P (C-layout) to per-wave LDS for re-read in A-layout
      const int r = g + row_off;
      myP[r * 32 + l16] = (_Float16)p0;
      myP[r * 32 + 16 + l16] = (_Float16)p1;
    }
    // DS ops are in-order per wave; fence so loads below see our stores
    asm volatile("s_wait_dscnt 0x0" ::: "memory");

    // P(16x32 f16) @ V(32x64) -> acc, 4 WMMAs
    const _Float16* pp = myP + l16 * 32 + (half16 ? 8 : 0);
    v16h pf = cat8(*(const v8h*)pp, *(const v8h*)(pp + 16));
#pragma unroll
    for (int nt = 0; nt < 4; ++nt) {
      const _Float16* vp = sVt + (nt * 16 + l16) * 32 + (half16 ? 8 : 0);
      v16h bf = cat8(*(const v8h*)vp, *(const v8h*)(vp + 16));
      acc[nt] = wmma16(pf, bf, acc[nt]);
    }
  }

  // normalize and write fp32 output [B*T, HS]
#pragma unroll
  for (int nt = 0; nt < 4; ++nt)
#pragma unroll
    for (int g = 0; g < 8; ++g) {
      const int row = bt0 + qbase + g + row_off;
      const int col = nt * 16 + l16;
      out[(size_t)row * HS + col] = acc[nt][g] / l_i[g];
    }
}

extern "C" void kernel_launch(void* const* d_in, const int* in_sizes, int n_in,
                              void* d_out, int out_size, void* d_ws,
                              size_t ws_size, hipStream_t stream) {
  const float* x = (const float*)d_in[0];
  const float* Wk = (const float*)d_in[1];
  const float* Wv = (const float*)d_in[2];
  float* out = (float*)d_out;

  _Float16* K16 = (_Float16*)d_ws;               // [BT, HS] f16 (2 MB)
  _Float16* V16 = K16 + (size_t)BT * HS;         // [BT, HS] f16 (2 MB)

  proj_kv<<<BT / 128, 256, 0, stream>>>(x, Wk, Wv, K16, V16);
  flash_attn<<<BDIM * (TDIM / 128), 256, 0, stream>>>(K16, V16, out);
}
